// Net_17669495456405
// MI455X (gfx1250) — matrix operand order repaired and checked by reading, not compiled, not run on hardware
//
#include <hip/hip_runtime.h>
#include <hip/hip_bf16.h>

typedef __attribute__((ext_vector_type(16))) _Float16 v16h;
typedef __attribute__((ext_vector_type(8)))  _Float16 v8h;
typedef __attribute__((ext_vector_type(8)))  float    v8f;

#define BN_EPS 1e-5f

__device__ inline void atomicMaxF(float* addr, float v) {
  // Works for all float values given init = -inf (0xFF800000)
  if (v >= 0.0f) atomicMax((int*)addr, __float_as_int(v));
  else           atomicMin((unsigned int*)addr, __float_as_uint(v));
}

// ---------------------------------------------------------------------------
// Generic fill
// ---------------------------------------------------------------------------
__global__ void k_fill(float* __restrict__ p, float v, long long n) {
  long long i = blockIdx.x * (long long)blockDim.x + threadIdx.x;
  if (i < n) p[i] = v;
}

// ---------------------------------------------------------------------------
// Self-loop attr: per-dst degree and mean incoming edge weight
// ---------------------------------------------------------------------------
__global__ void k_deg(const int* __restrict__ ei, const float* __restrict__ ew,
                      float* __restrict__ deg, float* __restrict__ loopw, int E) {
  int e = blockIdx.x * blockDim.x + threadIdx.x;
  if (e >= E) return;
  int d = ei[E + e];
  atomicAdd(&deg[d], 1.0f);
  atomicAdd(&loopw[d], ew[e]);
}

__global__ void k_loopattr(float* __restrict__ loopw, const float* __restrict__ deg, int N) {
  int i = blockIdx.x * blockDim.x + threadIdx.x;
  if (i >= N) return;
  loopw[i] = loopw[i] / fmaxf(deg[i], 1.0f);
}

// ---------------------------------------------------------------------------
// f32 -> padded f16 conversion for WMMA operands
//   Xh[i, 0:Kp] = f16(X[i, 0:K]) with zero pad K..Kp
// ---------------------------------------------------------------------------
__global__ void k_cvtA(const float* __restrict__ X, _Float16* __restrict__ Xh,
                       long long total, int K, int Kp) {
  long long idx = blockIdx.x * (long long)blockDim.x + threadIdx.x;
  if (idx >= total) return;
  int j = (int)(idx % Kp);
  long long i = idx / Kp;
  Xh[idx] = (j < K) ? (_Float16)X[i * K + j] : (_Float16)0.0f;
}

//   Wt[col, 0:Kp] = f16(W[0:K, col])  (transpose, zero pad K..Kp)
__global__ void k_cvtBt(const float* __restrict__ W, _Float16* __restrict__ Wt,
                        int K, int Kp, int M) {
  int idx = blockIdx.x * blockDim.x + threadIdx.x;
  if (idx >= M * Kp) return;
  int col = idx / Kp, k = idx % Kp;
  Wt[idx] = (k < K) ? (_Float16)W[(size_t)k * M + col] : (_Float16)0.0f;
}

// ---------------------------------------------------------------------------
// WMMA GEMM: Y[nrows,M] = Xh[nrows,KP] @ Wt[M,KP]^T   (f16 in, f32 out)
// One wave per 16x16 output tile; KP consumed in fully unrolled chunks of 32.
// REQUIRES nrows % 16 == 0 (true here: 50000, 512) -> unconditional stores.
// Fragment layouts used here (A/B use contiguous k, same order for both; not the ISA table order):
//   A (16x32 f16): lane half h, halves 0..15 = K kb+16h..kb+16h+15 (32B run)
//   B (32x16 f16): lane half h, halves 0..15 = K kb+16h..kb+16h+15 (32B run,
//                  contiguous because Wt is K-major per column)
//   D (16x16 f32): vgpr v -> row v+8h, col lane&15
// ---------------------------------------------------------------------------
template <int KP, int M>
__global__ void k_gemm_wmma(const _Float16* __restrict__ Xh,
                            const _Float16* __restrict__ Wt,
                            float* __restrict__ Y, int nrows) {
  int wave = threadIdx.x >> 5;
  int lane = threadIdx.x & 31;
  constexpr int tilesN = M >> 4;
  int tilesM = nrows >> 4;
  int tile = blockIdx.x * (blockDim.x >> 5) + wave;
  if (tile >= tilesM * tilesN) return;          // wave-uniform: EXEC stays all-1s
  int tm = tile / tilesN, tn = tile % tilesN;
  int half = lane >> 4, mr = lane & 15;
  int arow = (tm << 4) + mr;
  int bcol = (tn << 4) + mr;
  const _Float16* xp = Xh + (size_t)arow * KP;
  const _Float16* wp = Wt + (size_t)bcol * KP;

  v8f acc = {};
#pragma unroll
  for (int kb = 0; kb < KP; kb += 32) {
    v16h a = *(const v16h*)(xp + kb + 16 * half);
    v16h b = *(const v16h*)(wp + kb + 16 * half);
    acc = __builtin_amdgcn_wmma_f32_16x16x32_f16(
        /*neg_a=*/false, a, /*neg_b=*/false, b,
        /*c_mod=*/(short)0, acc, /*reuse_a=*/false, /*reuse_b=*/false);
  }
  // Unconditional stores; M compile-time => immediate offsets from one base.
  float* yp = Y + (size_t)((tm << 4) + 8 * half) * M + bcol;
#pragma unroll
  for (int v = 0; v < 8; ++v) yp[v * M] = acc[v];
}

// ---------------------------------------------------------------------------
// Per-node attention dots: hs = h.a_s, hd = h.a_d
// ---------------------------------------------------------------------------
__global__ void k_dots(const float* __restrict__ h, const float* __restrict__ as_,
                       const float* __restrict__ ad_, float* __restrict__ hs,
                       float* __restrict__ hd, int N, int M) {
  int i = blockIdx.x * blockDim.x + threadIdx.x;
  if (i >= N) return;
  float s1 = 0.0f, s2 = 0.0f;
  const float* row = h + (size_t)i * M;
  for (int j = 0; j < M; ++j) { float v = row[j]; s1 += v * as_[j]; s2 += v * ad_[j]; }
  hs[i] = s1;
  hd[i] = s2;
}

// edge-attr coefficient: c = dot(We[0,:], ae)   (ea @ We @ ae == ea * c)
__global__ void k_escal(const float* __restrict__ We, const float* __restrict__ ae,
                        int M, float* __restrict__ out) {
  float c = 0.0f;
  for (int j = 0; j < M; ++j) c += We[j] * ae[j];
  out[0] = c;
}

// ---------------------------------------------------------------------------
// Edge pass 1: alpha = leakyrelu(hs[src]+hd[dst]+ea*c), segment max into mmax
// ---------------------------------------------------------------------------
__global__ void k_alpha(const int* __restrict__ ei, const float* __restrict__ ew,
                        const float* __restrict__ loopw, const float* __restrict__ hs,
                        const float* __restrict__ hd, const float* __restrict__ escal,
                        float* __restrict__ alpha, float* __restrict__ mmax,
                        int E, int N) {
  long long e = blockIdx.x * (long long)blockDim.x + threadIdx.x;
  if (e >= (long long)E + N) return;
  int s, d; float w;
  if (e < E) { s = ei[e]; d = ei[E + e]; w = ew[e]; }
  else       { s = d = (int)(e - E);     w = loopw[s]; }
  float a = hs[s] + hd[d] + w * escal[0];
  a = (a >= 0.0f) ? a : 0.2f * a;
  alpha[e] = a;
  atomicMaxF(&mmax[d], a);
}

// Edge pass 2: exp(alpha - m[dst]), segment sum
__global__ void k_expsum(const int* __restrict__ ei, float* __restrict__ alpha,
                         const float* __restrict__ mmax, float* __restrict__ ssum,
                         int E, int N) {
  long long e = blockIdx.x * (long long)blockDim.x + threadIdx.x;
  if (e >= (long long)E + N) return;
  int d = (e < E) ? ei[E + e] : (int)(e - E);
  float ex = expf(alpha[e] - mmax[d]);
  alpha[e] = ex;
  atomicAdd(&ssum[d], ex);
}

// Edge pass 3: scatter att * h[src] into gat[dst]; one thread per (edge, feature)
__global__ void k_scatter(const int* __restrict__ ei, const float* __restrict__ alpha,
                          const float* __restrict__ ssum, const float* __restrict__ h,
                          float* __restrict__ gat, int E, int N, int M) {
  long long idx = blockIdx.x * (long long)blockDim.x + threadIdx.x;
  long long tot = ((long long)E + N) * M;
  if (idx >= tot) return;
  long long e = idx / M;
  int j = (int)(idx - e * M);
  int s, d;
  if (e < E) { s = ei[e]; d = ei[E + e]; }
  else       { s = d = (int)(e - E); }
  const float* hrow = h + (size_t)s * M;
  __builtin_prefetch(hrow, 0, 1);                         // global_prefetch_b8
  float att = alpha[e] / fmaxf(ssum[d], 1e-38f);
  atomicAdd(&gat[(size_t)d * M + j], att * hrow[j]);
}

// ---------------------------------------------------------------------------
// BatchNorm column stats (biased var), one block per column
// ---------------------------------------------------------------------------
__global__ void k_bnstats(const float* __restrict__ x, int n, int M,
                          float* __restrict__ mean, float* __restrict__ var) {
  __shared__ float s1[256];
  __shared__ float s2[256];
  int j = blockIdx.x;
  float a = 0.0f, b = 0.0f;
  for (int i = threadIdx.x; i < n; i += blockDim.x) {
    float v = x[(size_t)i * M + j];
    a += v; b += v * v;
  }
  s1[threadIdx.x] = a; s2[threadIdx.x] = b;
  __syncthreads();
  for (int s = 128; s > 0; s >>= 1) {
    if (threadIdx.x < s) { s1[threadIdx.x] += s1[threadIdx.x + s]; s2[threadIdx.x] += s2[threadIdx.x + s]; }
    __syncthreads();
  }
  if (threadIdx.x == 0) {
    float mu = s1[0] / n;
    mean[j] = mu;
    var[j]  = s2[0] / n - mu * mu;
  }
}

// Layer 1 epilogue: x1 = [ bn(gat32) | x16 ]
__global__ void k_layer1(const float* __restrict__ gat, const float* __restrict__ mean,
                         const float* __restrict__ var, const float* __restrict__ g,
                         const float* __restrict__ bb, const float* __restrict__ xin,
                         float* __restrict__ xout, int N) {
  int idx = blockIdx.x * blockDim.x + threadIdx.x;
  if (idx >= N * 48) return;
  int i = idx / 48, j = idx % 48;
  float y;
  if (j < 32) y = g[j] * (gat[(size_t)i * 32 + j] - mean[j]) * rsqrtf(var[j] + BN_EPS) + bb[j];
  else        y = xin[(size_t)i * 16 + (j - 32)];
  xout[idx] = y;
}

// Layers 2/3 epilogue: xout = bn(gat48) + xprev
__global__ void k_layer_res(const float* __restrict__ gat, const float* __restrict__ mean,
                            const float* __restrict__ var, const float* __restrict__ g,
                            const float* __restrict__ bb, const float* __restrict__ xprev,
                            float* __restrict__ xout, int N) {
  int idx = blockIdx.x * blockDim.x + threadIdx.x;
  if (idx >= N * 48) return;
  int j = idx % 48;
  float y = g[j] * (gat[idx] - mean[j]) * rsqrtf(var[j] + BN_EPS) + bb[j];
  xout[idx] = y + xprev[idx];
}

// ---------------------------------------------------------------------------
// Graph pooling (sum / max / count)
// ---------------------------------------------------------------------------
__global__ void k_pool(const float* __restrict__ x3, const int* __restrict__ batch,
                       float* __restrict__ padd, float* __restrict__ pmax,
                       float* __restrict__ pcnt, int N) {
  int idx = blockIdx.x * blockDim.x + threadIdx.x;
  if (idx >= N * 48) return;
  int i = idx / 48, j = idx % 48;
  int gb = batch[i];
  float v = x3[idx];
  atomicAdd(&padd[(size_t)gb * 48 + j], v);
  atomicMaxF(&pmax[(size_t)gb * 48 + j], v);
  if (j == 0) atomicAdd(&pcnt[gb], 1.0f);
}

__global__ void k_feat(const float* __restrict__ padd, const float* __restrict__ pmax,
                       const float* __restrict__ pcnt, float* __restrict__ feat, int G) {
  int idx = blockIdx.x * blockDim.x + threadIdx.x;
  if (idx >= G * 144) return;
  int gb = idx / 144, j = idx % 144;
  float v;
  if (j < 48)       v = padd[(size_t)gb * 48 + j];
  else if (j < 96)  v = pmax[(size_t)gb * 48 + (j - 48)];
  else              v = padd[(size_t)gb * 48 + (j - 96)] / fmaxf(pcnt[gb], 1.0f);
  feat[idx] = v;
}

// MLP bias + PReLU (in place)
__global__ void k_prelu(float* __restrict__ hmid, const float* __restrict__ bm1,
                        const float* __restrict__ prelu, int G) {
  int idx = blockIdx.x * blockDim.x + threadIdx.x;
  if (idx >= G * 144) return;
  int j = idx % 144;
  float v = hmid[idx] + bm1[j];
  hmid[idx] = (v >= 0.0f) ? v : prelu[0] * v;
}

// Final head: logits = hmid @ Wm2 + bm2, log_softmax over 2 classes
__global__ void k_head(const float* __restrict__ hmid, const float* __restrict__ Wm2,
                       const float* __restrict__ bm2, float* __restrict__ out, int G) {
  int gb = blockIdx.x * blockDim.x + threadIdx.x;
  if (gb >= G) return;
  float l0 = bm2[0], l1 = bm2[1];
  const float* row = hmid + (size_t)gb * 144;
  for (int j = 0; j < 144; ++j) {
    float v = row[j];
    l0 += v * Wm2[2 * j];
    l1 += v * Wm2[2 * j + 1];
  }
  float mx  = fmaxf(l0, l1);
  float lse = mx + logf(expf(l0 - mx) + expf(l1 - mx));
  out[2 * gb]     = l0 - lse;
  out[2 * gb + 1] = l1 - lse;
}

// ---------------------------------------------------------------------------
// Host orchestration
// ---------------------------------------------------------------------------
static inline int nb(long long n, int t) { return (int)((n + t - 1) / t); }

extern "C" void kernel_launch(void* const* d_in, const int* in_sizes, int n_in,
                              void* d_out, int out_size, void* d_ws, size_t ws_size,
                              hipStream_t stream) {
  (void)n_in; (void)ws_size;
  const float* x     = (const float*)d_in[0];
  const int*   ei    = (const int*)d_in[1];
  const float* ew    = (const float*)d_in[2];
  const int*   batch = (const int*)d_in[3];
  // per-layer params: W, asrc, adst, We, ae, b(no-op through BN), g, bb
  const float* Wp[3]  = {(const float*)d_in[4],  (const float*)d_in[12], (const float*)d_in[20]};
  const float* asp[3] = {(const float*)d_in[5],  (const float*)d_in[13], (const float*)d_in[21]};
  const float* adp[3] = {(const float*)d_in[6],  (const float*)d_in[14], (const float*)d_in[22]};
  const float* Wep[3] = {(const float*)d_in[7],  (const float*)d_in[15], (const float*)d_in[23]};
  const float* aep[3] = {(const float*)d_in[8],  (const float*)d_in[16], (const float*)d_in[24]};
  const float* gp[3]  = {(const float*)d_in[10], (const float*)d_in[18], (const float*)d_in[26]};
  const float* bbp[3] = {(const float*)d_in[11], (const float*)d_in[19], (const float*)d_in[27]};
  const float* Wm1   = (const float*)d_in[28];
  const float* bm1   = (const float*)d_in[29];
  const float* prelu = (const float*)d_in[30];
  const float* Wm2   = (const float*)d_in[31];
  const float* bm2   = (const float*)d_in[32];

  const int N = in_sizes[0] / 16;
  const int E = in_sizes[2];
  const int G = out_size / 2;
  const long long EP = (long long)E + N;

  // workspace carve-up (f32 region, then 32B-aligned f16 region)
  float* f = (float*)d_ws;
  float* deg   = f; f += N;
  float* loopw = f; f += N;
  float* hbuf  = f; f += (size_t)N * 48;
  float* hs    = f; f += N;
  float* hd    = f; f += N;
  float* alpha = f; f += EP;
  float* mmax  = f; f += N;
  float* ssum  = f; f += N;
  float* gat   = f; f += (size_t)N * 48;
  float* x1    = f; f += (size_t)N * 48;
  float* x2    = f; f += (size_t)N * 48;
  float* x3    = f; f += (size_t)N * 48;
  float* escal = f; f += 1;
  float* mean  = f; f += 48;
  float* var   = f; f += 48;
  float* padd  = f; f += (size_t)G * 48;
  float* pmax  = f; f += (size_t)G * 48;
  float* pcnt  = f; f += G;
  float* feat  = f; f += (size_t)G * 144;
  float* hmid  = f; f += (size_t)G * 144;
  f = (float*)(((uintptr_t)f + 31) & ~(uintptr_t)31);
  _Float16* xh = (_Float16*)f; f += (size_t)N * 32;      // N*64 halves (max Kp=64 path)
  _Float16* wt = (_Float16*)f; f += (144 * 160) / 2;     // max M*Kp halves

  // self-loop mean edge-attr
  k_fill<<<nb(2LL * N, 256), 256, 0, stream>>>(deg, 0.0f, 2LL * N);  // deg + loopw contiguous
  k_deg<<<nb(E, 256), 256, 0, stream>>>(ei, ew, deg, loopw, E);
  k_loopattr<<<nb(N, 256), 256, 0, stream>>>(loopw, deg, N);

  const float* Xin[3]  = {x, x1, x2};
  float*       Xout[3] = {x1, x2, x3};
  const int    Ks[3]   = {16, 48, 48};
  const int    Kps[3]  = {32, 64, 64};
  const int    Ms[3]   = {32, 48, 48};

  for (int l = 0; l < 3; ++l) {
    const int K = Ks[l], Kp = Kps[l], M = Ms[l];
    // convert operands to padded f16 (A row-major, B transposed K-major)
    k_cvtA<<<nb((long long)N * Kp, 256), 256, 0, stream>>>(Xin[l], xh, (long long)N * Kp, K, Kp);
    k_cvtBt<<<nb((long long)M * Kp, 256), 256, 0, stream>>>(Wp[l], wt, K, Kp, M);
    // h = Xin @ W  (WMMA)
    long long tiles = (long long)(N / 16) * (M / 16);
    if (l == 0) k_gemm_wmma<32, 32><<<nb(tiles, 8), 256, 0, stream>>>(xh, wt, hbuf, N);
    else        k_gemm_wmma<64, 48><<<nb(tiles, 8), 256, 0, stream>>>(xh, wt, hbuf, N);
    // attention precomputation
    k_dots<<<nb(N, 256), 256, 0, stream>>>(hbuf, asp[l], adp[l], hs, hd, N, M);
    k_escal<<<1, 1, 0, stream>>>(Wep[l], aep[l], M, escal);
    // init segment buffers
    k_fill<<<nb(N, 256), 256, 0, stream>>>(mmax, -__builtin_huge_valf(), N);
    k_fill<<<nb(N, 256), 256, 0, stream>>>(ssum, 0.0f, N);
    k_fill<<<nb((long long)N * M, 256), 256, 0, stream>>>(gat, 0.0f, (long long)N * M);
    // edge softmax + aggregate
    k_alpha<<<nb(EP, 256), 256, 0, stream>>>(ei, ew, loopw, hs, hd, escal, alpha, mmax, E, N);
    k_expsum<<<nb(EP, 256), 256, 0, stream>>>(ei, alpha, mmax, ssum, E, N);
    k_scatter<<<nb(EP * M, 256), 256, 0, stream>>>(ei, alpha, ssum, hbuf, gat, E, N, M);
    // batch norm + combine
    k_bnstats<<<M, 256, 0, stream>>>(gat, N, M, mean, var);
    if (l == 0)
      k_layer1<<<nb((long long)N * 48, 256), 256, 0, stream>>>(gat, mean, var, gp[l], bbp[l], x, x1, N);
    else
      k_layer_res<<<nb((long long)N * 48, 256), 256, 0, stream>>>(gat, mean, var, gp[l], bbp[l], Xin[l], Xout[l], N);
  }

  // graph pooling
  k_fill<<<nb((long long)G * 48, 256), 256, 0, stream>>>(padd, 0.0f, (long long)G * 48);
  k_fill<<<nb((long long)G * 48, 256), 256, 0, stream>>>(pmax, -__builtin_huge_valf(), (long long)G * 48);
  k_fill<<<nb(G, 256), 256, 0, stream>>>(pcnt, 0.0f, G);
  k_pool<<<nb((long long)N * 48, 256), 256, 0, stream>>>(x3, batch, padd, pmax, pcnt, N);
  k_feat<<<nb((long long)G * 144, 256), 256, 0, stream>>>(padd, pmax, pcnt, feat, G);

  // MLP head: feat @ Wm1 (WMMA, K=144 padded to 160) + bias + PReLU, then head
  k_cvtA<<<nb((long long)G * 160, 256), 256, 0, stream>>>(feat, xh, (long long)G * 160, 144, 160);
  k_cvtBt<<<nb(144 * 160, 256), 256, 0, stream>>>(Wm1, wt, 144, 160, 144);
  long long mtiles = (long long)(G / 16) * (144 / 16);
  k_gemm_wmma<160, 144><<<nb(mtiles, 8), 256, 0, stream>>>(xh, wt, hmid, G);
  k_prelu<<<nb((long long)G * 144, 256), 256, 0, stream>>>(hmid, bm1, prelu, G);
  k_head<<<nb(G, 128), 128, 0, stream>>>(hmid, Wm2, bm2, (float*)d_out, G);
}
